// IML_17592186044754
// MI455X (gfx1250) — matrix operand order repaired and checked by reading, not compiled, not run on hardware
//
#include <hip/hip_runtime.h>
#include <hip/hip_bf16.h>
#include <math.h>

typedef __attribute__((ext_vector_type(16))) _Float16 v16h;
typedef __attribute__((ext_vector_type(8)))  _Float16 v8h;
typedef __attribute__((ext_vector_type(8)))  float    v8f;

#define NTOK  65536   // b*h*w = 16*64*64
#define CDIM  256
#define KCODE 512
#define RATE      0.999f
#define EPS_NORM  1e-12f
#define EPS_MEAN  1e-6f

// ---------------- WMMA helpers (CDNA5 16x16x32 f16, f32 accum) ----------------
// A tile (16 tokens x 32 K): per ISA layout, lane r=lane&15 is row M=r.
// lanes 0-15 hold K = kbase+0..7 and kbase+16..23; lanes 16-31 hold K = +8..15, +24..31.
__device__ __forceinline__ v16h load_a16(const _Float16* base, int ldk, int kbase, int lane) {
  int r  = lane & 15;
  int k0 = kbase + ((lane >> 4) << 3);
  const _Float16* p = base + (size_t)r * ldk + k0;
  v8h lo = *(const v8h*)p;
  v8h hi = *(const v8h*)(p + 16);
  v16h o;
#pragma unroll
  for (int i = 0; i < 8; ++i) { o[i] = lo[i]; o[i + 8] = hi[i]; }
  return o;
}

// B tile (32 K x 16 N): lane n=lane&15 is column N=n.
// lanes 0-15 hold K = kbase+0..7 and kbase+16..23; lanes 16-31 hold K = +8..15, +24..31.
// Memory layout: base + n*ldk + k  (i.e. N-major rows, K contiguous).
__device__ __forceinline__ v16h load_b16(const _Float16* base, int ldk, int kbase, int lane) {
  int n  = lane & 15;
  int k0 = kbase + ((lane >> 4) << 3);
  const _Float16* p = base + (size_t)n * ldk + k0;
  v8h lo = *(const v8h*)p;
  v8h hi = *(const v8h*)(p + 16);
  v16h o;
#pragma unroll
  for (int i = 0; i < 8; ++i) { o[i] = lo[i]; o[i + 8] = hi[i]; }
  return o;
}

__device__ __forceinline__ v8f wmma_f16(v16h a, v16h b, v8f c) {
  return __builtin_amdgcn_wmma_f32_16x16x32_f16(false, a, false, b, (short)0, c, false, false);
}

// ---------------- K1: transpose x [b,c,h,w] -> token-major f16 + per-token norm ----
__global__ void prep_x(const float* __restrict__ x,
                       _Float16* __restrict__ xh, _Float16* __restrict__ xnh) {
  __shared__ _Float16 tile[64 * 256];   // [token(w)][ch ^ token] swizzled, 32KB
  __shared__ float psum[256];
  __shared__ float invn[64];
  int t  = threadIdx.x;
  int bb = blockIdx.x >> 6;   // batch 0..15
  int hh = blockIdx.x & 63;   // h     0..63
  const float* src = x + (size_t)bb * 256 * 4096 + (size_t)hh * 64;
#pragma unroll 4
  for (int s = 0; s < 64; ++s) {
    int flat = s * 256 + t;
    int ch = flat >> 6, w = flat & 63;       // 4 ch-rows of 64 contiguous floats per step
    float v = src[(size_t)ch * 4096 + w];
    tile[w * 256 + (ch ^ w)] = (_Float16)v;
  }
  __syncthreads();
  {
    int tk = t & 63, q = t >> 6;             // 4 partials per token
    float s0 = 0.f;
#pragma unroll 8
    for (int u = 0; u < 64; ++u) {
      int ch = q * 64 + u;
      float v = (float)tile[tk * 256 + (ch ^ tk)];
      s0 += v * v;
    }
    psum[q * 64 + tk] = s0;
  }
  __syncthreads();
  if (t < 64) {
    float s = psum[t] + psum[64 + t] + psum[128 + t] + psum[192 + t];
    invn[t] = 1.0f / fmaxf(sqrtf(s), EPS_NORM);
  }
  __syncthreads();
  size_t n0 = (size_t)blockIdx.x * 64;
#pragma unroll 4
  for (int s = 0; s < 64; ++s) {
    _Float16 v = tile[s * 256 + (t ^ s)];
    xh [(n0 + s) * 256 + t] = v;
    xnh[(n0 + s) * 256 + t] = (_Float16)((float)v * invn[s]);
  }
}

// ---------------- K2: codebook prep: mf copy, mnh (normalized f16), mhT f16 ----
__global__ void prep_m(const float* __restrict__ m, float* __restrict__ mf,
                       _Float16* __restrict__ mnh, _Float16* __restrict__ mhT) {
  __shared__ float red[256];
  int code = blockIdx.x, t = threadIdx.x;
  float v = m[(size_t)code * 256 + t];
  mf[(size_t)code * 256 + t] = v;
  red[t] = v * v;
  __syncthreads();
  for (int s = 128; s > 0; s >>= 1) { if (t < s) red[t] += red[t + s]; __syncthreads(); }
  float inv = 1.0f / fmaxf(sqrtf(red[0]), EPS_NORM);
  mnh[(size_t)code * 256 + t] = (_Float16)(v * inv);
  mhT[(size_t)t * 512 + code] = (_Float16)v;
}

// ---------------- zero stats ----------------
__global__ void zero_f32(float* __restrict__ p, int n) {
  int i = blockIdx.x * 256 + threadIdx.x;
  if (i < n) p[i] = 0.0f;
}

// ---------------- K3: fused score->argmax->counts/embed_sum (no score buffer) ----
__global__ void assign_kernel(const _Float16* __restrict__ xnh, const _Float16* __restrict__ xh,
                              const _Float16* __restrict__ mnh,
                              float* __restrict__ counts, float* __restrict__ esum) {
  int lane = threadIdx.x & 31;
  int wave = threadIdx.x >> 5;
  int tok0 = (blockIdx.x * 8 + wave) * 16;     // 16 tokens per wave
  const _Float16* arow = xnh + (size_t)tok0 * CDIM;

  float rmax[8]; int ridx[8];
#pragma unroll
  for (int j = 0; j < 8; ++j) { rmax[j] = -1e30f; ridx[j] = 0; }

  for (int ct = 0; ct < KCODE / 16; ++ct) {            // 32 code tiles
    v8f acc = {};
    const _Float16* brow = mnh + (size_t)ct * 16 * CDIM;
#pragma unroll
    for (int ks = 0; ks < CDIM / 32; ++ks) {           // 8 K-steps of 32
      v16h A = load_a16(arow, CDIM, ks * 32, lane);
      v16h B = load_b16(brow, CDIM, ks * 32, lane);
      acc = wmma_f16(A, B, acc);
    }
    int code = ct * 16 + (lane & 15);
#pragma unroll
    for (int j = 0; j < 8; ++j) {
      float v = acc[j];
      if (v > rmax[j]) { rmax[j] = v; ridx[j] = code; }   // strict > keeps first max
    }
  }
  // cross-lane argmax within each 16-lane half (rows j in lanes 0-15, j+8 in 16-31)
#pragma unroll
  for (int j = 0; j < 8; ++j) {
    float mx = rmax[j]; int id = ridx[j];
#pragma unroll
    for (int msk = 1; msk < 16; msk <<= 1) {
      float omx = __shfl_xor(mx, msk, 32);
      int   oid = __shfl_xor(id, msk, 32);
      if (omx > mx || (omx == mx && oid < id)) { mx = omx; id = oid; }
    }
    ridx[j] = id;
  }
  int ids[16];
#pragma unroll
  for (int j = 0; j < 8; ++j) {
    ids[j]     = __shfl(ridx[j], 0, 32);    // token tok0+j
    ids[j + 8] = __shfl(ridx[j], 16, 32);   // token tok0+8+j
  }
  if (lane == 0) {
#pragma unroll
    for (int t = 0; t < 16; ++t) atomicAdd(&counts[ids[t]], 1.0f);
  }
  // embed_sum scatter: 32 lanes cover 256 channels (8 each)
#pragma unroll
  for (int t = 0; t < 16; ++t) {
    const v8h xv = *(const v8h*)(xh + (size_t)(tok0 + t) * CDIM + lane * 8);
    float* dst = esum + (size_t)ids[t] * CDIM + lane * 8;
#pragma unroll
    for (int u = 0; u < 8; ++u) atomicAdd(dst + u, (float)xv[u]);
  }
}

// ---------------- K4: EMA update + renormalize ----------------
__global__ void update_m(const float* __restrict__ counts, const float* __restrict__ esum,
                         float* __restrict__ mf, _Float16* __restrict__ mnh,
                         _Float16* __restrict__ mhT) {
  __shared__ float red[256];
  int code = blockIdx.x, t = threadIdx.x;
  float cnt  = counts[code];
  float mean = esum[(size_t)code * 256 + t] / (cnt + EPS_MEAN);
  float v    = mf[(size_t)code * 256 + t] * RATE + mean * (1.0f - RATE);
  mf[(size_t)code * 256 + t] = v;
  red[t] = v * v;
  __syncthreads();
  for (int s = 128; s > 0; s >>= 1) { if (t < s) red[t] += red[t + s]; __syncthreads(); }
  float inv = 1.0f / fmaxf(sqrtf(red[0]), EPS_NORM);
  mnh[(size_t)code * 256 + t] = (_Float16)(v * inv);
  mhT[(size_t)t * 512 + code] = (_Float16)v;
}

// ---------------- K5: score -> LDS -> softmax -> (P @ m) via WMMA ----------------
__global__ void __launch_bounds__(128) final_kernel(const _Float16* __restrict__ xnh,
                                                    const _Float16* __restrict__ mnh,
                                                    const _Float16* __restrict__ mhT,
                                                    float* __restrict__ outws) {
  __shared__ __align__(16) _Float16 sc[4 * 16 * 512];  // 64KB: per wave 16x512 scores/probs
  int lane = threadIdx.x & 31;
  int wave = threadIdx.x >> 5;             // 4 waves/block
  int tok0 = (blockIdx.x * 4 + wave) * 16;
  _Float16* myrow = &sc[(size_t)wave * 16 * 512];
  const _Float16* arow = xnh + (size_t)tok0 * CDIM;

  // phase 1: scores 16x512 -> LDS (f16)
  for (int ct = 0; ct < KCODE / 16; ++ct) {
    v8f acc = {};
    const _Float16* brow = mnh + (size_t)ct * 16 * CDIM;
#pragma unroll
    for (int ks = 0; ks < CDIM / 32; ++ks) {
      v16h A = load_a16(arow, CDIM, ks * 32, lane);
      v16h B = load_b16(brow, CDIM, ks * 32, lane);
      acc = wmma_f16(A, B, acc);
    }
    int r = lane & 15, rowadd = (lane >> 4) * 8;
#pragma unroll
    for (int j = 0; j < 8; ++j)
      myrow[(j + rowadd) * 512 + ct * 16 + r] = (_Float16)acc[j];
  }
  asm volatile("s_wait_dscnt 0" ::: "memory");

  // phase 2: softmax in-place in LDS; 2 lanes per token (each handles 256 entries)
  {
    int tk = lane >> 1, hf = lane & 1;
    _Float16* srow = myrow + tk * 512 + hf * 256;
    float mx = -1e30f;
#pragma unroll 4
    for (int i = 0; i < 256; i += 8) {
      v8h v = *(const v8h*)(srow + i);
#pragma unroll
      for (int u = 0; u < 8; ++u) mx = fmaxf(mx, (float)v[u]);
    }
    mx = fmaxf(mx, __shfl_xor(mx, 1, 32));
    float sum = 0.f;
#pragma unroll 4
    for (int i = 0; i < 256; i += 8) {
      v8h v = *(const v8h*)(srow + i);
#pragma unroll
      for (int u = 0; u < 8; ++u) sum += expf((float)v[u] - mx);
    }
    sum += __shfl_xor(sum, 1, 32);
    float inv = 1.0f / sum;
#pragma unroll 4
    for (int i = 0; i < 256; i += 8) {
      v8h v = *(const v8h*)(srow + i);
      v8h p;
#pragma unroll
      for (int u = 0; u < 8; ++u) p[u] = (_Float16)(expf((float)v[u] - mx) * inv);
      *(v8h*)(srow + i) = p;
    }
  }
  asm volatile("s_wait_dscnt 0" ::: "memory");

  // phase 3: out[16 x 256] = P(16x512, in LDS) @ m (via mhT[ch][code])
  for (int ct2 = 0; ct2 < CDIM / 16; ++ct2) {
    v8f acc = {};
    const _Float16* brow = mhT + (size_t)ct2 * 16 * KCODE;
#pragma unroll
    for (int ks = 0; ks < KCODE / 32; ++ks) {          // 16 K-steps
      v16h A = load_a16(myrow, 512, ks * 32, lane);    // ds_read path
      v16h B = load_b16(brow, KCODE, ks * 32, lane);
      acc = wmma_f16(A, B, acc);
    }
    int r = lane & 15, rowadd = (lane >> 4) * 8;
#pragma unroll
    for (int j = 0; j < 8; ++j)
      outws[(size_t)(tok0 + j + rowadd) * CDIM + ct2 * 16 + r] = acc[j];
  }
}

// ---------------- K6: transpose [N,c] f32 back to [b,c,h,w] ----------------
__global__ void unprep_out(const float* __restrict__ outws, float* __restrict__ out) {
  __shared__ float tile[64 * 256];   // 64KB, XOR-swizzled
  int t  = threadIdx.x;
  int bb = blockIdx.x >> 6;
  int hh = blockIdx.x & 63;
  size_t n0 = (size_t)blockIdx.x * 64;
#pragma unroll 4
  for (int s = 0; s < 64; ++s)
    tile[s * 256 + (t ^ s)] = outws[(n0 + s) * 256 + t];
  __syncthreads();
  float* dst = out + (size_t)bb * 256 * 4096 + (size_t)hh * 64;
#pragma unroll 4
  for (int s = 0; s < 64; ++s) {
    int flat = s * 256 + t;
    int ch = flat >> 6, w = flat & 63;
    dst[(size_t)ch * 4096 + w] = tile[w * 256 + (ch ^ w)];
  }
}

// ---------------- host launch ----------------
extern "C" void kernel_launch(void* const* d_in, const int* in_sizes, int n_in,
                              void* d_out, int out_size, void* d_ws, size_t ws_size,
                              hipStream_t stream) {
  const float* x = (const float*)d_in[0];   // [16,256,64,64]
  const float* m = (const float*)d_in[1];   // [512,256]
  float* out = (float*)d_out;
  char* ws = (char*)d_ws;

  // workspace layout (bytes)
  _Float16* xh     = (_Float16*)(ws + 0);          // 33,554,432  token f16 [N,256]
  _Float16* xnh    = (_Float16*)(ws + 33554432);   // 33,554,432  normalized tokens f16
  _Float16* mnh    = (_Float16*)(ws + 67108864);   //    262,144  normalized codebook f16 [k,c]
  _Float16* mhT    = (_Float16*)(ws + 67371008);   //    262,144  codebook f16 transposed [c,k]
  float*    mf     = (float*)   (ws + 67633152);   //    524,288  working codebook f32
  float*    counts = (float*)   (ws + 68157440);   //      2,048
  float*    esum   = (float*)   (ws + 68159488);   //    524,288  (contiguous after counts)
  float*    outws  = (float*)   (ws + 68683776);   // 67,108,864  out tokens f32 [N,256]

  prep_x<<<1024, 256, 0, stream>>>(x, xh, xnh);
  prep_m<<<512, 256, 0, stream>>>(m, mf, mnh, mhT);
  for (int it = 0; it < 3; ++it) {
    zero_f32<<<(512 + 512 * 256 + 255) / 256, 256, 0, stream>>>(counts, 512 + 512 * 256);
    assign_kernel<<<NTOK / 128, 256, 0, stream>>>(xnh, xh, mnh, counts, esum);
    update_m<<<512, 256, 0, stream>>>(counts, esum, mf, mnh, mhT);
  }
  final_kernel<<<NTOK / 64, 128, 0, stream>>>(xnh, mnh, mhT, outws);
  unprep_out<<<1024, 256, 0, stream>>>(outws, out);
}
